// GraphAttentionLayer_15650860827430
// MI455X (gfx1250) — matrix operand order repaired and checked by reading, not compiled, not run on hardware
//
#include <hip/hip_runtime.h>

typedef __attribute__((ext_vector_type(16))) _Float16 v16h;
typedef __attribute__((ext_vector_type(8)))  _Float16 v8h;
typedef __attribute__((ext_vector_type(8)))  float    v8f;

#define GAT_N 8192
#define GAT_F 256
#define GAT_ALPHA 0.2f

// LDS: f2 tile (32KB) + 16x8192 f16 e/weight tile (256KB) + reduction scratch
#define SMEM_GAT ((GAT_N * 4) + (16 * GAT_N * 2) + (16 + 512 + 512 + 16 + 16) * 4)

#define WMMA_F16(A, B, C) __builtin_amdgcn_wmma_f32_16x16x32_f16( \
    false, (A), false, (B), (short)0, (C), false, false)

// ------------------------------------------------------------------
// Kernel 1: Wh = h @ W, stored f16 column-major WhhT[F][N]
// (column-major makes phase-2 B fragments a single contiguous 32B load)
// ------------------------------------------------------------------
__global__ __launch_bounds__(32)
void k_wh(const float* __restrict__ h, const float* __restrict__ W,
          _Float16* __restrict__ whhT) {
  const int lane = threadIdx.x;
  const int half = lane >> 4;
  const int mr   = lane & 15;
  const int mt = blockIdx.y;   // 0..511 (row tile)
  const int nt = blockIdx.x;   // 0..15  (col tile)
  const int n = nt * 16 + mr;  // B col for this lane
  const float* hrow = h + (size_t)(mt * 16 + mr) * GAT_F;

  v8f acc = {};
  for (int k0 = 0; k0 < GAT_F; k0 += 32) {
    // A 16-bit layout: two contiguous 8-float runs at k0+half*8 and k0+16+half*8
    const float4 a01 = *(const float4*)(hrow + k0 + half * 8);
    const float4 a23 = *(const float4*)(hrow + k0 + half * 8 + 4);
    const float4 a45 = *(const float4*)(hrow + k0 + 16 + half * 8);
    const float4 a67 = *(const float4*)(hrow + k0 + 16 + half * 8 + 4);
    v16h a, b;
    a[0]=(_Float16)a01.x; a[1]=(_Float16)a01.y; a[2]=(_Float16)a01.z; a[3]=(_Float16)a01.w;
    a[4]=(_Float16)a23.x; a[5]=(_Float16)a23.y; a[6]=(_Float16)a23.z; a[7]=(_Float16)a23.w;
    a[8]=(_Float16)a45.x; a[9]=(_Float16)a45.y; a[10]=(_Float16)a45.z; a[11]=(_Float16)a45.w;
    a[12]=(_Float16)a67.x; a[13]=(_Float16)a67.y; a[14]=(_Float16)a67.z; a[15]=(_Float16)a67.w;
    #pragma unroll
    for (int j = 0; j < 16; ++j) {
      // B 16-bit layout: table order, K = half*8 + (j&7) + 16*(j>>3) (column gather)
      const int kb = k0 + half * 8 + (j & 7) + ((j >> 3) << 4);
      b[j] = (_Float16)W[(size_t)kb * GAT_F + n];
    }
    acc = WMMA_F16(a, b, acc);
  }
  #pragma unroll
  for (int r = 0; r < 8; ++r) {
    const int row = mt * 16 + r + half * 8;
    const int col = nt * 16 + mr;
    whhT[(size_t)col * GAT_N + row] = (_Float16)acc[r];
  }
}

// ------------------------------------------------------------------
// Kernel 2: f1 = Wh@a1, f2 = Wh@a2 (one wave per row, shuffle reduce)
// ------------------------------------------------------------------
__global__ __launch_bounds__(256)
void k_f12(const _Float16* __restrict__ whhT, const float* __restrict__ a,
           float* __restrict__ f1, float* __restrict__ f2) {
  const int wave = threadIdx.x >> 5;
  const int lane = threadIdx.x & 31;
  const int row  = blockIdx.x * 8 + wave;
  float s1 = 0.f, s2 = 0.f;
  for (int c = lane; c < GAT_F; c += 32) {
    const float v = (float)whhT[(size_t)c * GAT_N + row];
    s1 += v * a[c];
    s2 += v * a[GAT_F + c];
  }
  #pragma unroll
  for (int off = 16; off; off >>= 1) {
    s1 += __shfl_xor(s1, off, 32);
    s2 += __shfl_xor(s2, off, 32);
  }
  if (lane == 0) { f1[row] = s1; f2[row] = s2; }
}

// ------------------------------------------------------------------
// Kernel 3: fused masked-softmax attention + attention@Wh + ELU.
// One block = 16 output rows, 512 threads (16 waves).
//  Phase 1  : stream adj once from HBM (b128 + prefetch), e tile -> LDS f16,
//             online (max,sum) per row.
//  Phase 1.5: convert e tile in place to softmax weights (exp once, no
//             per-wave duplication).
//  Phase 2  : wave w owns n-tile w; inner loop = 2 ds_load_b128 +
//             2 global_load_b128 + WMMA, dual accumulators.
// ------------------------------------------------------------------
__global__ __launch_bounds__(512)
void k_gat(const int* __restrict__ adj, const float* __restrict__ f1g,
           const float* __restrict__ f2g, const _Float16* __restrict__ whhT,
           float* __restrict__ out) {
  extern __shared__ char smem[];
  float*    f2s  = (float*)smem;                 // 8192 floats
  _Float16* e16  = (_Float16*)(f2s + GAT_N);     // 16 x 8192 halves
  float*    f1s  = (float*)(e16 + 16 * GAT_N);   // 16
  float*    redm = f1s + 16;                     // 512
  float*    reds = redm + 512;                   // 512
  float*    mrow = reds + 512;                   // 16
  float*    sinv = mrow + 16;                    // 16

  const int tid  = threadIdx.x;
  const int row0 = blockIdx.x * 16;
  const int r = tid >> 5;           // row 0..15, one wave per row
  const int q = tid & 31;

  for (int i = tid; i < GAT_N; i += 512) f2s[i] = f2g[i];
  if (tid < 16) f1s[tid] = f1g[row0 + tid];
  __syncthreads();

  // ---------------- Phase 1: e tile + online softmax stats ----------------
  {
    const float f1v = f1s[r];
    const int4* arow = (const int4*)(adj + (size_t)(row0 + r) * GAT_N);
    _Float16* erow = e16 + r * GAT_N;
    float m = -3.0e38f, s = 0.0f;
    for (int it = 0; it < GAT_N / 128; ++it) {
      const int idx = it * 32 + q;
      __builtin_prefetch(arow + idx + 256, 0, 3);   // global_prefetch ~4KB ahead
      const int4 av = arow[idx];
      const int j = idx * 4;
      const float4 fv = *(const float4*)(f2s + j);
      union { _Float16 hh[4]; unsigned long long u; } pk;
      #pragma unroll
      for (int d = 0; d < 4; ++d) {
        const int   ad  = (d == 0) ? av.x : (d == 1) ? av.y : (d == 2) ? av.z : av.w;
        const float f2v = (d == 0) ? fv.x : (d == 1) ? fv.y : (d == 2) ? fv.z : fv.w;
        float e = f1v + f2v;
        e = fmaxf(e, GAT_ALPHA * e);               // leaky_relu, branchless
        if (ad > 0) {
          if (e <= m) { s += __expf(e - m); }
          else        { s = s * __expf(m - e) + 1.0f; m = e; }
          pk.hh[d] = (_Float16)e;
        } else {
          pk.hh[d] = (_Float16)(-30000.0f);        // exp -> 0 after subtracting max
        }
      }
      *(unsigned long long*)(erow + j) = pk.u;
    }
    redm[tid] = m;
    reds[tid] = s;
  }
  __syncthreads();
  if (tid < 16) {   // combine 32 partial (max,sum) pairs per row
    float m = -3.0e38f, s = 0.f;
    for (int qq = 0; qq < 32; ++qq) {
      const float mq = redm[tid * 32 + qq], sq = reds[tid * 32 + qq];
      const float M = fmaxf(m, mq);
      s = s * __expf(m - M) + sq * __expf(mq - M);
      m = M;
    }
    mrow[tid] = fmaxf(m, -29000.0f);   // clamp: empty row -> weights 0, no NaN
    sinv[tid] = (s > 0.f) ? (1.0f / s) : 0.f;
  }
  __syncthreads();

  // ------- Phase 1.5: e -> softmax weight, in place, computed once -------
  {
    const float mi = mrow[r];
    const float si = sinv[r];
    _Float16* erow = e16 + r * GAT_N;
    for (int it = 0; it < GAT_N / 256; ++it) {     // 32 iters of v8h
      const int j = (it * 32 + q) * 8;
      v8h ev = *(v8h*)(erow + j);
      v8h wv;
      #pragma unroll
      for (int d = 0; d < 8; ++d)
        wv[d] = (_Float16)(__expf((float)ev[d] - mi) * si);
      *(v8h*)(erow + j) = wv;
    }
  }
  __syncthreads();

  // ---------------- Phase 2: WMMA attention @ Wh ----------------
  const int wave = tid >> 5;        // wave w owns n-tile w
  const int lane = tid & 31;
  const int half = lane >> 4;
  const int mr   = lane & 15;
  const _Float16* wrow = e16 + mr * GAT_N;                       // weights
  const _Float16* bcol = whhT + (size_t)(wave * 16 + mr) * GAT_N;

  v8f acc0 = {}, acc1 = {};
  for (int c = 0; c < GAT_N; c += 64) {
    const v8h w0 = *(const v8h*)(wrow + c + half * 16);
    const v8h w1 = *(const v8h*)(wrow + c + half * 16 + 8);
    const v16h a0 = __builtin_shufflevector(w0, w1,
        0,1,2,3,4,5,6,7,8,9,10,11,12,13,14,15);
    const v16h b0 = *(const v16h*)(bcol + c + half * 16);

    const v8h w2 = *(const v8h*)(wrow + c + 32 + half * 16);
    const v8h w3 = *(const v8h*)(wrow + c + 40 + half * 16);
    const v16h a1 = __builtin_shufflevector(w2, w3,
        0,1,2,3,4,5,6,7,8,9,10,11,12,13,14,15);
    const v16h b1 = *(const v16h*)(bcol + c + 32 + half * 16);

    acc0 = WMMA_F16(a0, b0, acc0);   // independent chains: acc0 / acc1
    acc1 = WMMA_F16(a1, b1, acc1);
  }

  // Epilogue: combine accumulators, fused ELU, write f32 output
  #pragma unroll
  for (int rr = 0; rr < 8; ++rr) {
    const int row = row0 + rr + half * 8;
    const int c0  = wave * 16 + mr;
    float x = acc0[rr] + acc1[rr];
    x = (x > 0.f) ? x : (__expf(x) - 1.0f);
    out[(size_t)row * GAT_F + c0] = x;
  }
}

// ------------------------------------------------------------------
extern "C" void kernel_launch(void* const* d_in, const int* in_sizes, int n_in,
                              void* d_out, int out_size, void* d_ws, size_t ws_size,
                              hipStream_t stream) {
  const float* h   = (const float*)d_in[0];
  const int*   adj = (const int*)  d_in[1];
  const float* W   = (const float*)d_in[2];
  const float* a   = (const float*)d_in[3];
  float* out = (float*)d_out;

  // workspace: WhhT f16 [F][N] (4MB) | f1 f32 [N] | f2 f32 [N]
  _Float16* whhT = (_Float16*)d_ws;
  float* f1 = (float*)((char*)d_ws + (size_t)GAT_N * GAT_F * sizeof(_Float16));
  float* f2 = f1 + GAT_N;

  (void)hipFuncSetAttribute(reinterpret_cast<const void*>(k_gat),
                            hipFuncAttributeMaxDynamicSharedMemorySize, SMEM_GAT);

  k_wh <<<dim3(GAT_F / 16, GAT_N / 16), 32,  0,        stream>>>(h, W, whhT);
  k_f12<<<dim3(GAT_N / 8),              256, 0,        stream>>>(whhT, a, f1, f2);
  k_gat<<<dim3(GAT_N / 16),             512, SMEM_GAT, stream>>>(adj, f1, f2, whhT, out);
}
